// SSMKernel_53077205844148
// MI455X (gfx1250) — matrix operand order repaired and checked by reading, not compiled, not run on hardware
//
#include <hip/hip_runtime.h>
#include <hip/hip_bf16.h>
#include <math.h>

// ---------------------------------------------------------------------------
// Mamba SSM block forward for gfx1250 (MI455X).
//   B=1, L=4096, D_MODEL=512, D_INNER=1024, D_STATE=16
// GEMMs on v_wmma_f32_16x16x32_bf16 (bf16 in, f32 accumulate).
// Recurrence via chunked 2-level scan (32 chunks x 128 steps), f32 state.
// ---------------------------------------------------------------------------

#define L_SEQ     4096
#define DMODEL    512
#define DINNER    1024
#define DST       16
#define NPROJ     33       // 1 + 16 + 16
#define NCHUNK    32
#define CHLEN     128      // NCHUNK * CHLEN == L_SEQ

typedef __bf16 bf16_t;
typedef bf16_t bf16x8  __attribute__((ext_vector_type(8)));
typedef bf16_t bf16x16 __attribute__((ext_vector_type(16)));
typedef float  f32x8   __attribute__((ext_vector_type(8)));

// ------------------------------- helpers -----------------------------------

__device__ __forceinline__ float silu_f(float x) {
  return x / (1.0f + __expf(-x));
}
__device__ __forceinline__ float softplus_f(float x) {
  // numerically stable softplus
  return fmaxf(x, 0.0f) + log1pf(__expf(-fabsf(x)));
}

// --------------------------- f32 -> bf16 cast ------------------------------

__global__ __launch_bounds__(256) void cast_f32_bf16_kernel(
    const float* __restrict__ in, bf16_t* __restrict__ out, int n) {
  int i = blockIdx.x * 256 + threadIdx.x;
  if (i < n) out[i] = (bf16_t)in[i];
}

// ------------------------------ WMMA GEMM ----------------------------------
// C[M,N] = A[M,K] * W[N,K]^T, all row-major; A,W bf16; C f32.
// Wave computes a 16x64 strip: 4 accumulators reuse one A fragment.

__device__ __forceinline__ bf16x16 load_a_frag(
    const bf16_t* __restrict__ A, int lda, int m0, int k0, int lane) {
  // 16x32 bf16 A tile, ISA layout:
  //   lanes 0-15 : row M=lane,    VGPR0-3 = K0..7,  VGPR4-7 = K16..23
  //   lanes16-31 : row M=lane-16, VGPR0-3 = K8..15, VGPR4-7 = K24..31
  const int r  = lane & 15;
  const int kh = (lane >> 4) << 3;           // 0 or 8
  const bf16_t* p = A + (size_t)(m0 + r) * lda + (k0 + kh);
  bf16x8 lo = *(const bf16x8*)p;             // K = kh .. kh+7
  bf16x8 hi = *(const bf16x8*)(p + 16);      // K = kh+16 .. kh+23
  return __builtin_shufflevector(lo, hi, 0,1,2,3,4,5,6,7,8,9,10,11,12,13,14,15);
}

__device__ __forceinline__ bf16x16 load_b_frag(
    const bf16_t* __restrict__ W, int ldw, int n0, int k0, int lane) {
  // 32x16 bf16 B tile with B[k][n] = W[n][k]:
  //   lanes 0-15 : col N=lane,    K = k0..k0+7,  k0+16..k0+23
  //   lanes16-31 : col N=lane-16, K = k0+8..k0+15, k0+24..k0+31
  const int n  = n0 + (lane & 15);
  const int kh = (lane >> 4) << 3;           // 0 or 8
  const bf16_t* p = W + (size_t)n * ldw + (k0 + kh);
  bf16x8 lo = *(const bf16x8*)p;
  bf16x8 hi = *(const bf16x8*)(p + 16);
  return __builtin_shufflevector(lo, hi, 0,1,2,3,4,5,6,7,8,9,10,11,12,13,14,15);
}

__global__ __launch_bounds__(256) void wmma_gemm_bt_kernel(
    const bf16_t* __restrict__ A, const bf16_t* __restrict__ W,
    float* __restrict__ C, int M, int N, int K) {
  const int lane    = threadIdx.x & 31;
  const int nGroups = N >> 6;                         // 64 cols per wave
  const int waveId  = blockIdx.x * 8 + (threadIdx.x >> 5);
  const int mTile   = waveId / nGroups;
  const int nGroup  = waveId - mTile * nGroups;
  const int m0      = mTile << 4;
  if (m0 >= M) return;                                // wave-uniform
  const int n0      = nGroup << 6;

  f32x8 acc0 = {}, acc1 = {}, acc2 = {}, acc3 = {};
  for (int k0 = 0; k0 < K; k0 += 32) {
    bf16x16 a  = load_a_frag(A, K, m0, k0, lane);
    bf16x16 b0 = load_b_frag(W, K, n0 +  0, k0, lane);
    bf16x16 b1 = load_b_frag(W, K, n0 + 16, k0, lane);
    bf16x16 b2 = load_b_frag(W, K, n0 + 32, k0, lane);
    bf16x16 b3 = load_b_frag(W, K, n0 + 48, k0, lane);
    acc0 = __builtin_amdgcn_wmma_f32_16x16x32_bf16(false, a, false, b0, (short)0, acc0, false, false);
    acc1 = __builtin_amdgcn_wmma_f32_16x16x32_bf16(false, a, false, b1, (short)0, acc1, false, false);
    acc2 = __builtin_amdgcn_wmma_f32_16x16x32_bf16(false, a, false, b2, (short)0, acc2, false, false);
    acc3 = __builtin_amdgcn_wmma_f32_16x16x32_bf16(false, a, false, b3, (short)0, acc3, false, false);
  }

  // C/D layout: lanes 0-15 -> N=lane, rows m0..m0+7 in elems 0..7;
  //             lanes16-31 -> N=lane-16, rows m0+8..m0+15.
  const int nc = n0 + (lane & 15);
  const int mb = m0 + ((lane >> 4) << 3);
  #pragma unroll
  for (int r = 0; r < 8; ++r) {
    float* crow = C + (size_t)(mb + r) * N + nc;
    crow[ 0] = acc0[r];
    crow[16] = acc1[r];
    crow[32] = acc2[r];
    crow[48] = acc3[r];
  }
}

// ------------------------ depthwise conv(3) + SiLU -------------------------
// xc[l,d] = silu( sum_k xz[l-1+k, d] * conv_w[d,k] + conv_b[d] ), x_in = xz[:, :1024]

__global__ __launch_bounds__(256) void conv_silu_kernel(
    const float* __restrict__ xz, const float* __restrict__ conv_w,
    const float* __restrict__ conv_b, float* __restrict__ xc) {
  int i = blockIdx.x * 256 + threadIdx.x;        // over L*DINNER
  int d = i & (DINNER - 1);
  int l = i >> 10;
  float acc = conv_b[d];
  #pragma unroll
  for (int k = 0; k < 3; ++k) {
    int ll = l + k - 1;
    if (ll >= 0 && ll < L_SEQ)
      acc = fmaf(xz[(size_t)ll * (2 * DINNER) + d], conv_w[d * 3 + k], acc);
  }
  xc[(size_t)l * DINNER + d] = silu_f(acc);
}

// ------------------------------- x_proj ------------------------------------
// dbc[l, 0:33] = xc[l, :] @ x_proj_w[33,1024]^T  (too skinny for WMMA: N=33)

__global__ __launch_bounds__(256) void xproj_kernel(
    const float* __restrict__ xc, const float* __restrict__ xpw,
    float* __restrict__ dbc) {
  __shared__ float srow[DINNER];
  const int l = blockIdx.x;
  for (int i = threadIdx.x; i < DINNER; i += 256)
    srow[i] = xc[(size_t)l * DINNER + i];
  __syncthreads();
  const int lane = threadIdx.x & 31;
  const int wave = threadIdx.x >> 5;
  for (int e = wave; e < NPROJ; e += 8) {
    const float* w = xpw + (size_t)e * DINNER;
    float sum = 0.0f;
    for (int i = lane; i < DINNER; i += 32) sum = fmaf(srow[i], w[i], sum);
    #pragma unroll
    for (int off = 16; off > 0; off >>= 1) sum += __shfl_down(sum, off, 32);
    if (lane == 0) dbc[(size_t)l * NPROJ + e] = sum;
  }
}

// ----------------------- delta = softplus(raw*dtw+dtb) ---------------------

__global__ __launch_bounds__(256) void delta_kernel(
    const float* __restrict__ dbc, const float* __restrict__ dt_w,
    const float* __restrict__ dt_b, float* __restrict__ delta) {
  int i = blockIdx.x * 256 + threadIdx.x;        // over L*DINNER
  int d = i & (DINNER - 1);
  int l = i >> 10;
  float x = fmaf(dbc[(size_t)l * NPROJ], dt_w[d], dt_b[d]);
  delta[i] = softplus_f(x);
}

// ----------------------- scan phase 1: per-chunk summary -------------------
// For chunk c, channel d: ap[s] = prod(exp(dt*A)), h[s] = chunk scan from 0.

__global__ __launch_bounds__(256) void scan_phase1_kernel(
    const float* __restrict__ delta, const float* __restrict__ xc,
    const float* __restrict__ dbc,   const float* __restrict__ A_log,
    float* __restrict__ P, float* __restrict__ H) {
  const int d = blockIdx.x * 256 + threadIdx.x;  // 0..1023
  const int c = blockIdx.y;                      // 0..NCHUNK-1
  float Ad[DST], ap[DST], h[DST];
  #pragma unroll
  for (int s = 0; s < DST; ++s) {
    Ad[s] = -__expf(A_log[d * DST + s]);
    ap[s] = 1.0f;
    h[s]  = 0.0f;
  }
  const int l0 = c * CHLEN;
  for (int li = 0; li < CHLEN; ++li) {
    const int l = l0 + li;
    const float dt = delta[(size_t)l * DINNER + d];
    const float u  = dt * xc[(size_t)l * DINNER + d];
    const float* brow = dbc + (size_t)l * NPROJ + 1;
    #pragma unroll
    for (int s = 0; s < DST; ++s) {
      float a = __expf(dt * Ad[s]);
      h[s]  = fmaf(a, h[s], u * brow[s]);
      ap[s] *= a;
    }
  }
  const size_t base = ((size_t)c * DINNER + d) * DST;
  #pragma unroll
  for (int s = 0; s < DST; ++s) { P[base + s] = ap[s]; H[base + s] = h[s]; }
}

// ----------------------- scan phase 2: cross-chunk prefix ------------------

__global__ __launch_bounds__(256) void scan_phase2_kernel(
    const float* __restrict__ P, const float* __restrict__ H,
    float* __restrict__ Hinit) {
  const int t = blockIdx.x * 256 + threadIdx.x;  // 0..16383
  const int d = t >> 4;
  const int s = t & 15;
  float carry = 0.0f;
  for (int c = 0; c < NCHUNK; ++c) {
    const size_t idx = ((size_t)c * DINNER + d) * DST + s;
    Hinit[idx] = carry;                          // exclusive prefix state
    carry = fmaf(P[idx], carry, H[idx]);
  }
}

// ------ scan phase 3: re-scan with init state, fuse C-dot, D, SiLU(z) ------

__global__ __launch_bounds__(256) void scan_phase3_kernel(
    const float* __restrict__ delta, const float* __restrict__ xc,
    const float* __restrict__ dbc,   const float* __restrict__ A_log,
    const float* __restrict__ Hinit, const float* __restrict__ xz,
    const float* __restrict__ D_param, float* __restrict__ ycore) {
  const int d = blockIdx.x * 256 + threadIdx.x;
  const int c = blockIdx.y;
  float Ad[DST], h[DST];
  const size_t base = ((size_t)c * DINNER + d) * DST;
  #pragma unroll
  for (int s = 0; s < DST; ++s) {
    Ad[s] = -__expf(A_log[d * DST + s]);
    h[s]  = Hinit[base + s];
  }
  const float Dp = D_param[d];
  const int l0 = c * CHLEN;
  for (int li = 0; li < CHLEN; ++li) {
    const int l = l0 + li;
    const float dt  = delta[(size_t)l * DINNER + d];
    const float xcv = xc[(size_t)l * DINNER + d];
    const float u   = dt * xcv;
    const float* brow = dbc + (size_t)l * NPROJ + 1;
    const float* crow = brow + DST;
    float yv = 0.0f;
    #pragma unroll
    for (int s = 0; s < DST; ++s) {
      float a = __expf(dt * Ad[s]);
      h[s] = fmaf(a, h[s], u * brow[s]);
      yv   = fmaf(h[s], crow[s], yv);
    }
    yv += xcv * Dp;
    const float zv = xz[(size_t)l * (2 * DINNER) + DINNER + d];
    yv *= silu_f(zv);
    ycore[(size_t)l * DINNER + d] = yv;          // safe alias of delta[l,d]
  }
}

// ------------------------------ layernorm ----------------------------------

__global__ __launch_bounds__(256) void layernorm_kernel(
    const float* __restrict__ y, const float* __restrict__ g,
    const float* __restrict__ b, bf16_t* __restrict__ yln) {
  __shared__ float s1[256], s2[256];
  const int l   = blockIdx.x;
  const int tid = threadIdx.x;
  const float* row = y + (size_t)l * DINNER;
  float v[4], sum = 0.0f, sq = 0.0f;
  #pragma unroll
  for (int j = 0; j < 4; ++j) {
    v[j] = row[tid + 256 * j];
    sum += v[j];
    sq   = fmaf(v[j], v[j], sq);
  }
  s1[tid] = sum; s2[tid] = sq;
  __syncthreads();
  for (int off = 128; off > 0; off >>= 1) {
    if (tid < off) { s1[tid] += s1[tid + off]; s2[tid] += s2[tid + off]; }
    __syncthreads();
  }
  if (tid == 0) {
    const float mu  = s1[0] * (1.0f / DINNER);
    const float var = s2[0] * (1.0f / DINNER) - mu * mu;
    s1[0] = mu;
    s2[0] = rsqrtf(var + 1e-5f);
  }
  __syncthreads();
  const float mu = s1[0], rstd = s2[0];
  #pragma unroll
  for (int j = 0; j < 4; ++j) {
    const int d = tid + 256 * j;
    yln[(size_t)l * DINNER + d] = (bf16_t)((v[j] - mu) * rstd * g[d] + b[d]);
  }
}

// ------------------------------ launcher -----------------------------------

extern "C" void kernel_launch(void* const* d_in, const int* in_sizes, int n_in,
                              void* d_out, int out_size, void* d_ws, size_t ws_size,
                              hipStream_t stream) {
  const float* x         = (const float*)d_in[0];
  const float* in_proj_w = (const float*)d_in[1];
  const float* x_proj_w  = (const float*)d_in[2];
  const float* dt_w      = (const float*)d_in[3];
  const float* dt_b      = (const float*)d_in[4];
  const float* conv_w    = (const float*)d_in[5];
  const float* conv_b    = (const float*)d_in[6];
  const float* A_log     = (const float*)d_in[7];
  const float* D_param   = (const float*)d_in[8];
  const float* out_w     = (const float*)d_in[9];
  const float* ln_g      = (const float*)d_in[10];
  const float* ln_b      = (const float*)d_in[11];
  float* out = (float*)d_out;

  char* ws = (char*)d_ws;
  size_t off = 0;
  auto take = [&](size_t bytes) -> char* {
    char* p = ws + off;
    off += (bytes + 255) & ~(size_t)255;
    return p;
  };

  bf16_t* xb    = (bf16_t*)take((size_t)L_SEQ * DMODEL * 2);        //  4 MB
  bf16_t* winb  = (bf16_t*)take((size_t)2 * DINNER * DMODEL * 2);   //  2 MB
  bf16_t* woutb = (bf16_t*)take((size_t)DMODEL * DINNER * 2);       //  1 MB
  float*  xz    = (float*)take((size_t)L_SEQ * 2 * DINNER * 4);     // 32 MB
  float*  xc    = (float*)take((size_t)L_SEQ * DINNER * 4);         // 16 MB
  float*  dbc   = (float*)take((size_t)L_SEQ * NPROJ * 4);          // .5 MB
  float*  delta = (float*)take((size_t)L_SEQ * DINNER * 4);         // 16 MB
  float*  P     = (float*)take((size_t)NCHUNK * DINNER * DST * 4);  //  2 MB
  float*  H     = (float*)take((size_t)NCHUNK * DINNER * DST * 4);  //  2 MB
  float*  Hin   = (float*)take((size_t)NCHUNK * DINNER * DST * 4);  //  2 MB
  bf16_t* ylnb  = (bf16_t*)take((size_t)L_SEQ * DINNER * 2);        //  8 MB
  float*  ycore = delta;  // safe alias: phase3 reads delta[l,d] then writes
                          // ycore[l,d] in the same thread, same element.

  // 1) casts to bf16 for WMMA
  cast_f32_bf16_kernel<<<(L_SEQ * DMODEL + 255) / 256, 256, 0, stream>>>(
      x, xb, L_SEQ * DMODEL);
  cast_f32_bf16_kernel<<<(2 * DINNER * DMODEL + 255) / 256, 256, 0, stream>>>(
      in_proj_w, winb, 2 * DINNER * DMODEL);
  cast_f32_bf16_kernel<<<(DMODEL * DINNER + 255) / 256, 256, 0, stream>>>(
      out_w, woutb, DMODEL * DINNER);

  // 2) xz = x @ in_proj_w^T   (4096 x 2048, K=512)
  {
    int waves = (L_SEQ / 16) * ((2 * DINNER) / 64);   // 8192
    wmma_gemm_bt_kernel<<<waves / 8, 256, 0, stream>>>(
        xb, winb, xz, L_SEQ, 2 * DINNER, DMODEL);
  }

  // 3) depthwise conv + SiLU
  conv_silu_kernel<<<(L_SEQ * DINNER) / 256, 256, 0, stream>>>(
      xz, conv_w, conv_b, xc);

  // 4) dbc = xc @ x_proj_w^T
  xproj_kernel<<<L_SEQ, 256, 0, stream>>>(xc, x_proj_w, dbc);

  // 5) delta = softplus(dbc[:,0] * dt_w + dt_b)
  delta_kernel<<<(L_SEQ * DINNER) / 256, 256, 0, stream>>>(
      dbc, dt_w, dt_b, delta);

  // 6-8) chunked scan
  {
    dim3 grid(DINNER / 256, NCHUNK);
    scan_phase1_kernel<<<grid, 256, 0, stream>>>(delta, xc, dbc, A_log, P, H);
    scan_phase2_kernel<<<(DINNER * DST) / 256, 256, 0, stream>>>(P, H, Hin);
    scan_phase3_kernel<<<grid, 256, 0, stream>>>(
        delta, xc, dbc, A_log, Hin, xz, D_param, ycore);
  }

  // 9) layernorm -> bf16
  layernorm_kernel<<<L_SEQ, 256, 0, stream>>>(ycore, ln_g, ln_b, ylnb);

  // 10) out = y_ln @ out_w^T  (4096 x 512, K=1024)
  {
    int waves = (L_SEQ / 16) * (DMODEL / 64);         // 2048
    wmma_gemm_bt_kernel<<<waves / 8, 256, 0, stream>>>(
        ylnb, woutb, out, L_SEQ, DMODEL, DINNER);
  }

  (void)in_sizes; (void)n_in; (void)out_size; (void)ws_size;
}
